// RoPEALiBiTransformerEncoder_39453569581365
// MI455X (gfx1250) — matrix operand order repaired and checked by reading, not compiled, not run on hardware
//
#include <hip/hip_runtime.h>
#include <math.h>

typedef __attribute__((ext_vector_type(16))) __bf16 v16bf;
typedef __attribute__((ext_vector_type(8)))  __bf16 v8bf;
typedef __attribute__((ext_vector_type(8)))  float  v8f;

namespace {
constexpr int kNL = 8;
constexpr int kD  = 1024;
constexpr int kH  = 16;
constexpr int kHD = 64;
constexpr int kFF = 4096;
constexpr int kL  = 1024;
constexpr int kB  = 2;
constexpr int kM  = kB * kL;   // 2048 flattened rows
constexpr float kEps = 1e-5f;

// A-fragment (16x32 bf16, MxK): lanes 0-15 / 16-31 hold K-halves.
// VGPR0-7 = K[kh*16 .. kh*16+15] (contiguous).
__device__ inline v16bf load_frag_a(const __bf16* row, int kh) {
  v8bf lo = *(const v8bf*)(row + kh * 16);
  v8bf hi = *(const v8bf*)(row + kh * 16 + 8);
  return __builtin_shufflevector(lo, hi, 0,1,2,3,4,5,6,7,8,9,10,11,12,13,14,15);
}
// B-fragment (32x16 bf16, KxN) staged N-major: 16 contiguous K values.
__device__ inline v16bf load_frag_b(const __bf16* p) {
  v8bf lo = *(const v8bf*)(p);
  v8bf hi = *(const v8bf*)(p + 8);
  return __builtin_shufflevector(lo, hi, 0,1,2,3,4,5,6,7,8,9,10,11,12,13,14,15);
}
__device__ inline v8f wmma_bf16(v16bf a, v16bf b, v8f c) {
  return __builtin_amdgcn_wmma_f32_16x16x32_bf16(
      /*neg_a=*/false, a, /*neg_b=*/false, b,
      /*c_mod=*/(short)0, c, /*reuse_a=*/false, /*reuse_b=*/false);
}
} // namespace

// ---------------------------------------------------------------------------
// C[M,N] = act(A[M,K] @ W[K,N] + bias[N]); ACT: 0=none, 1=exact GELU
// Block: 128 threads = 4 waves; 64x64 tile; BK=32.
// ---------------------------------------------------------------------------
template <int ACT>
__global__ __launch_bounds__(128) void gemm_wmma_kernel(
    const float* __restrict__ A, const float* __restrict__ W,
    const float* __restrict__ bias, float* __restrict__ C,
    int M, int N, int K) {
  __shared__ __align__(16) __bf16 As[64][32];   // [m][k]
  __shared__ __align__(16) __bf16 Bs[64][32];   // [n][k] (transposed)
  const int tid  = threadIdx.x;
  const int lane = tid & 31;
  const int wave = tid >> 5;
  const int rowBase = blockIdx.y * 64;
  const int colBase = blockIdx.x * 64;
  const int m16 = lane & 15;
  const int kh  = lane >> 4;

  v8f acc[4] = {};

  for (int k0 = 0; k0 < K; k0 += 32) {
#pragma unroll
    for (int i = 0; i < 4; ++i) {           // stage A 64x32
      int idx = tid + i * 128;              // float4 units
      int r = idx >> 3;
      int c = (idx & 7) << 2;
      float4 f = *(const float4*)(A + (size_t)(rowBase + r) * K + k0 + c);
      As[r][c + 0] = (__bf16)f.x; As[r][c + 1] = (__bf16)f.y;
      As[r][c + 2] = (__bf16)f.z; As[r][c + 3] = (__bf16)f.w;
    }
#pragma unroll
    for (int i = 0; i < 4; ++i) {           // stage W 32x64 -> transposed
      int idx = tid + i * 128;
      int r = idx >> 4;                     // k
      int c = (idx & 15) << 2;              // n
      float4 f = *(const float4*)(W + (size_t)(k0 + r) * N + colBase + c);
      Bs[c + 0][r] = (__bf16)f.x; Bs[c + 1][r] = (__bf16)f.y;
      Bs[c + 2][r] = (__bf16)f.z; Bs[c + 3][r] = (__bf16)f.w;
    }
    __syncthreads();
    v16bf a = load_frag_a(&As[wave * 16 + m16][0], kh);
#pragma unroll
    for (int nt = 0; nt < 4; ++nt) {
      v16bf b = load_frag_b(&Bs[nt * 16 + m16][kh * 16]);
      acc[nt] = wmma_bf16(a, b, acc[nt]);
    }
    __syncthreads();
  }

  const int m0 = kh * 8;                    // C/D layout: m = 8*(lane/16)+r
#pragma unroll
  for (int nt = 0; nt < 4; ++nt) {
    int col = colBase + nt * 16 + m16;      // n = lane % 16
    float bv = bias[col];
#pragma unroll
    for (int r = 0; r < 8; ++r) {
      float v = acc[nt][r] + bv;
      if (ACT == 1) v = 0.5f * v * (1.0f + erff(v * 0.70710678118654752f));
      int row = rowBase + wave * 16 + m0 + r;
      C[(size_t)row * N + col] = v;
    }
  }
}

// ---------------------------------------------------------------------------
// RoPE exactly as reference (pe = concat(cos,sin); c=pe[::2], s=pe[1::2]),
// with [B,L,H,HD] -> [B,H,L,HD] transpose. One thread per (b,pos,h,j<32).
// ---------------------------------------------------------------------------
__global__ __launch_bounds__(256) void rope_kernel(
    const float* __restrict__ in, float* __restrict__ out) {
  int t = blockIdx.x * 256 + threadIdx.x;
  int j   = t & 31;
  int h   = (t >> 5) & (kH - 1);
  int pos = (t >> 9) & (kL - 1);
  int b   = t >> 19;
  const float* xr = in + ((size_t)(b * kL + pos) * kH + h) * kHD;
  float xe = xr[2 * j];
  float xo = xr[2 * j + 1];
  int iA = (j < 16) ? (2 * j) : (2 * (j - 16));
  float angA = (float)pos * powf(10000.0f, -(2.0f * (float)iA) / (float)kHD);
  float angB = (float)pos * powf(10000.0f, -(2.0f * (float)(iA + 1)) / (float)kHD);
  float cj = (j < 16) ? cosf(angA) : sinf(angA);
  float sj = (j < 16) ? cosf(angB) : sinf(angB);
  float* orow = out + ((size_t)(b * kH + h) * kL + pos) * kHD;
  orow[j]      = xe * cj - xo * sj;
  orow[32 + j] = xe * sj + xo * cj;
}

// ---------------------------------------------------------------------------
// Flash-style attention: one (b,h,64-query tile) per 128-thread block.
// q,k: [B,H,L,HD]; v,o: [B,L,H,HD]. S and P@V both via bf16 WMMA.
// ---------------------------------------------------------------------------
__global__ __launch_bounds__(128) void attn_wmma_kernel(
    const float* __restrict__ q, const float* __restrict__ k,
    const float* __restrict__ v, float* __restrict__ o) {
  __shared__ __align__(16) __bf16 Qs[64][64];   // [qrow][hd]
  __shared__ __align__(16) __bf16 Ks[64][64];   // [key][hd]
  __shared__ __align__(16) __bf16 Vt[64][64];   // [hd][key]
  __shared__ __align__(16) __bf16 Ps[64][64];   // softmax probs
  __shared__ float Ss[64][64];
  __shared__ float rowM[64], rowL[64], rowA[64];

  const int tid  = threadIdx.x;
  const int lane = tid & 31;
  const int wave = tid >> 5;
  const int qBase = blockIdx.x * 64;
  const int bh = blockIdx.y;
  const int h  = bh & (kH - 1);
  const int b  = bh >> 4;
  const int m16 = lane & 15;
  const int kh  = lane >> 4;
  const float scale = 0.125f;                       // HD^-0.5, HD=64
  const float slope = exp2f(-exp2f(-(log2f((float)kH) - 3.0f)) * (float)h);

  // stage Q (64x64) once
  const float* qp = q + ((size_t)bh * kL + qBase) * kHD;
#pragma unroll
  for (int i = 0; i < 8; ++i) {
    int idx = tid + i * 128;                 // float4 units, 1024 total
    int r = idx >> 4;
    int c = (idx & 15) << 2;
    float4 f = *(const float4*)(qp + (size_t)r * kHD + c);
    Qs[r][c + 0] = (__bf16)f.x; Qs[r][c + 1] = (__bf16)f.y;
    Qs[r][c + 2] = (__bf16)f.z; Qs[r][c + 3] = (__bf16)f.w;
  }
  if (tid < 64) { rowM[tid] = -1e30f; rowL[tid] = 0.0f; }
  v8f accO[4] = {};
  __syncthreads();

  for (int kt = 0; kt < kL; kt += 64) {
    // stage K (row-major) and V (transposed)
    const float* kp = k + ((size_t)bh * kL + kt) * kHD;
#pragma unroll
    for (int i = 0; i < 8; ++i) {
      int idx = tid + i * 128;
      int r = idx >> 4;
      int c = (idx & 15) << 2;
      float4 f = *(const float4*)(kp + (size_t)r * kHD + c);
      Ks[r][c + 0] = (__bf16)f.x; Ks[r][c + 1] = (__bf16)f.y;
      Ks[r][c + 2] = (__bf16)f.z; Ks[r][c + 3] = (__bf16)f.w;
      float4 g = *(const float4*)(v + (((size_t)(b * kL + kt + r)) * kH + h) * kHD + c);
      Vt[c + 0][r] = (__bf16)g.x; Vt[c + 1][r] = (__bf16)g.y;
      Vt[c + 2][r] = (__bf16)g.z; Vt[c + 3][r] = (__bf16)g.w;
    }
    __syncthreads();

    // S = Q @ K^T over HD=64 (two K=32 chunks), 4 key sub-tiles per wave
    v8f accS[4] = {};
#pragma unroll
    for (int kc = 0; kc < 2; ++kc) {
      v16bf a = load_frag_a(&Qs[wave * 16 + m16][kc * 32], kh);
#pragma unroll
      for (int nt = 0; nt < 4; ++nt) {
        v16bf bf = load_frag_b(&Ks[nt * 16 + m16][kc * 32 + kh * 16]);
        accS[nt] = wmma_bf16(a, bf, accS[nt]);
      }
    }
    {
      const int m0 = kh * 8;
#pragma unroll
      for (int nt = 0; nt < 4; ++nt)
#pragma unroll
        for (int r = 0; r < 8; ++r) {
          int row = wave * 16 + m0 + r;
          int col = nt * 16 + m16;
          float dist = (float)((qBase + row) - (kt + col));
          dist = dist > 0.0f ? dist : 0.0f;
          Ss[row][col] = accS[nt][r] * scale + slope * dist;
        }
    }
    __syncthreads();

    // online softmax (one row per thread, threads 0..63)
    if (tid < 64) {
      const int row = tid;
      float oldM = rowM[row];
      float tmax = -1e30f;
      for (int j = 0; j < 64; ++j) tmax = fmaxf(tmax, Ss[row][j]);
      float newM = fmaxf(oldM, tmax);
      float a = expf(oldM - newM);
      float sum = 0.0f;
      for (int j = 0; j < 64; ++j) {
        float e = expf(Ss[row][j] - newM);
        sum += e;
        Ps[row][j] = (__bf16)e;
      }
      rowL[row] = rowL[row] * a + sum;
      rowM[row] = newM;
      rowA[row] = a;
    }
    __syncthreads();

    // O = O*alpha + P @ V
    {
      const int m0 = kh * 8;
#pragma unroll
      for (int nt = 0; nt < 4; ++nt)
#pragma unroll
        for (int r = 0; r < 8; ++r)
          accO[nt][r] *= rowA[wave * 16 + m0 + r];
#pragma unroll
      for (int kc = 0; kc < 2; ++kc) {
        v16bf a = load_frag_a(&Ps[wave * 16 + m16][kc * 32], kh);
#pragma unroll
        for (int nt = 0; nt < 4; ++nt) {
          v16bf bf = load_frag_b(&Vt[nt * 16 + m16][kc * 32 + kh * 16]);
          accO[nt] = wmma_bf16(a, bf, accO[nt]);
        }
      }
    }
    __syncthreads();
  }

  // normalize and write o[b][qBase+row][h][hd]
  {
    const int m0 = kh * 8;
#pragma unroll
    for (int nt = 0; nt < 4; ++nt)
#pragma unroll
      for (int r = 0; r < 8; ++r) {
        int row = wave * 16 + m0 + r;
        int hd = nt * 16 + m16;
        float val = accO[nt][r] / rowL[row];
        o[(((size_t)(b * kL + qBase + row)) * kH + h) * kHD + hd] = val;
      }
  }
}

// ---------------------------------------------------------------------------
// out = LayerNorm(x + y) * g + b   (y may be nullptr). One row per block.
// ---------------------------------------------------------------------------
__global__ __launch_bounds__(256) void add_ln_kernel(
    const float* __restrict__ x, const float* __restrict__ y,
    const float* __restrict__ g, const float* __restrict__ b,
    float* __restrict__ out) {
  __shared__ float red[256];
  const int row = blockIdx.x;
  const int tid = threadIdx.x;
  const float* xr = x + (size_t)row * kD;
  float vals[4];
  float s = 0.0f;
#pragma unroll
  for (int i = 0; i < 4; ++i) {
    int c = tid + i * 256;
    float v = xr[c];
    if (y) v += y[(size_t)row * kD + c];
    vals[i] = v;
    s += v;
  }
  red[tid] = s; __syncthreads();
  for (int off = 128; off > 0; off >>= 1) {
    if (tid < off) red[tid] += red[tid + off];
    __syncthreads();
  }
  float mu = red[0] * (1.0f / kD);
  __syncthreads();
  float s2 = 0.0f;
#pragma unroll
  for (int i = 0; i < 4; ++i) { float d = vals[i] - mu; s2 += d * d; }
  red[tid] = s2; __syncthreads();
  for (int off = 128; off > 0; off >>= 1) {
    if (tid < off) red[tid] += red[tid + off];
    __syncthreads();
  }
  float rstd = rsqrtf(red[0] * (1.0f / kD) + kEps);
#pragma unroll
  for (int i = 0; i < 4; ++i) {
    int c = tid + i * 256;
    out[(size_t)row * kD + c] = (vals[i] - mu) * rstd * g[c] + b[c];
  }
}

// ---------------------------------------------------------------------------
extern "C" void kernel_launch(void* const* d_in, const int* in_sizes, int n_in,
                              void* d_out, int out_size, void* d_ws, size_t ws_size,
                              hipStream_t stream) {
  (void)in_sizes; (void)n_in; (void)out_size; (void)ws_size;
  const float* src = (const float*)d_in[0];
  const float* Wq  = (const float*)d_in[1];
  const float* bq  = (const float*)d_in[2];
  const float* Wk  = (const float*)d_in[3];
  const float* bk  = (const float*)d_in[4];
  const float* Wv  = (const float*)d_in[5];
  const float* bv  = (const float*)d_in[6];
  const float* Wo  = (const float*)d_in[7];
  const float* bo  = (const float*)d_in[8];
  const float* W1  = (const float*)d_in[9];
  const float* b1  = (const float*)d_in[10];
  const float* W2  = (const float*)d_in[11];
  const float* b2  = (const float*)d_in[12];
  const float* g1  = (const float*)d_in[13];
  const float* be1 = (const float*)d_in[14];
  const float* g2  = (const float*)d_in[15];
  const float* be2 = (const float*)d_in[16];
  const float* gf  = (const float*)d_in[17];
  const float* bfp = (const float*)d_in[18];

  float* ws = (float*)d_ws;
  const size_t NX = (size_t)kM * kD;   // 2M floats per activation buffer
  float* bufX  = ws;
  float* bufQ  = bufX  + NX;
  float* bufK  = bufQ  + NX;
  float* bufV  = bufK  + NX;
  float* bufQr = bufV  + NX;
  float* bufKr = bufQr + NX;
  float* bufO  = bufKr + NX;
  float* bufY  = bufO  + NX;
  float* bufFF = bufY  + NX;           // 2048 x 4096

  hipMemcpyAsync(bufX, src, NX * sizeof(float), hipMemcpyDeviceToDevice, stream);

  const dim3 blk128(128);
  const dim3 gD(kD / 64, kM / 64);     // N=1024 GEMMs
  const dim3 gFF(kFF / 64, kM / 64);   // N=4096 GEMM
  const int ropeBlocks = (kB * kL * kH * 32) / 256;

  for (int l = 0; l < kNL; ++l) {
    const float* wq = Wq + (size_t)l * kD * kD;
    const float* wk = Wk + (size_t)l * kD * kD;
    const float* wv = Wv + (size_t)l * kD * kD;
    const float* wo = Wo + (size_t)l * kD * kD;
    const float* w1 = W1 + (size_t)l * kD * kFF;
    const float* w2 = W2 + (size_t)l * kFF * kD;

    gemm_wmma_kernel<0><<<gD, blk128, 0, stream>>>(bufX, wq, bq + (size_t)l * kD, bufQ, kM, kD, kD);
    gemm_wmma_kernel<0><<<gD, blk128, 0, stream>>>(bufX, wk, bk + (size_t)l * kD, bufK, kM, kD, kD);
    gemm_wmma_kernel<0><<<gD, blk128, 0, stream>>>(bufX, wv, bv + (size_t)l * kD, bufV, kM, kD, kD);

    rope_kernel<<<ropeBlocks, 256, 0, stream>>>(bufQ, bufQr);
    rope_kernel<<<ropeBlocks, 256, 0, stream>>>(bufK, bufKr);

    attn_wmma_kernel<<<dim3(kL / 64, kB * kH), blk128, 0, stream>>>(bufQr, bufKr, bufV, bufO);

    gemm_wmma_kernel<0><<<gD, blk128, 0, stream>>>(bufO, wo, bo + (size_t)l * kD, bufY, kM, kD, kD);
    add_ln_kernel<<<kM, 256, 0, stream>>>(bufX, bufY, g1 + (size_t)l * kD, be1 + (size_t)l * kD, bufX);

    gemm_wmma_kernel<1><<<gFF, blk128, 0, stream>>>(bufX, w1, b1 + (size_t)l * kFF, bufFF, kM, kFF, kD);
    gemm_wmma_kernel<0><<<gD, blk128, 0, stream>>>(bufFF, w2, b2 + (size_t)l * kD, bufY, kM, kD, kFF);
    add_ln_kernel<<<kM, 256, 0, stream>>>(bufX, bufY, g2 + (size_t)l * kD, be2 + (size_t)l * kD, bufX);
  }

  add_ln_kernel<<<kM, 256, 0, stream>>>(bufX, nullptr, gf, bfp, (float*)d_out);
}
